// EnergyTransformer_26439818674791
// MI455X (gfx1250) — matrix operand order repaired and checked by reading, not compiled, not run on hardware
//
#include <hip/hip_runtime.h>
#include <hip/hip_bf16.h>

typedef __bf16 bf16_t;
typedef __attribute__((ext_vector_type(16))) __bf16 v16bf;
typedef __attribute__((ext_vector_type(8)))  __bf16 v8bf;
typedef __attribute__((ext_vector_type(8)))  float  v8f;

#define D_      768
#define H_      12
#define HEAD_   64
#define HID_    3072
#define L_      1024
#define NB_     4
#define BL_     4096
#define NIT_    12
#define ALPHA_  0.1f
#define EPS_    1e-5f

// ---------------- WMMA helpers (CDNA5 gfx1250, wave32) ----------------

__device__ __forceinline__ v8f wmma_bf16(v16bf a, v16bf b, v8f c) {
  // D = A(16x32 bf16) * B(32x16 bf16) + C(16x16 f32)
  return __builtin_amdgcn_wmma_f32_16x16x32_bf16(false, a, false, b, (short)0, c,
                                                 false, false);
}

__device__ __forceinline__ void zero8(v8f& c) {
#pragma unroll
  for (int i = 0; i < 8; ++i) c[i] = 0.f;
}

// A fragment: 16x32 tile of row-major A.  contiguous k order (same as B):
// lane(m=lane&15, hi=lane>>4), element e: K = hi*16 + e
__device__ __forceinline__ v16bf load_afrag(const bf16_t* A, int lda, int row0,
                                            int k0, int lane) {
  int m = lane & 15, hi = lane >> 4;
  const bf16_t* p = A + (size_t)(row0 + m) * lda + k0 + hi * 16;
  v8bf lo  = *(const v8bf*)(p);
  v8bf hi8 = *(const v8bf*)(p + 8);
  v16bf a;
#pragma unroll
  for (int i = 0; i < 8; ++i) { a[i] = lo[i]; a[i + 8] = hi8[i]; }
  return a;
}

// B fragment: 32x16 tile, operand stored [N,K] row-major ("NT" gemm).
// lane(n=lane&15, hi=lane>>4), element e: K = hi*16 + e  -> contiguous 32B load.
__device__ __forceinline__ v16bf load_bfrag(const bf16_t* B, int ldb, int col0,
                                            int k0, int lane) {
  int n = lane & 15, hi = lane >> 4;
  return *(const v16bf*)(B + (size_t)(col0 + n) * ldb + k0 + hi * 16);
}

// 64x64 wave tile: acc += A[row0:+64, :K] * B[col0:+64, :K]^T
__device__ __forceinline__ void gemm64(const bf16_t* A, int lda, const bf16_t* B,
                                       int ldb, int row0, int col0, int K,
                                       int lane, v8f (&acc)[4][4]) {
  for (int k0 = 0; k0 < K; k0 += 32) {
    v16bf a[4];
#pragma unroll
    for (int i = 0; i < 4; ++i) a[i] = load_afrag(A, lda, row0 + 16 * i, k0, lane);
#pragma unroll
    for (int j = 0; j < 4; ++j) {
      v16bf b = load_bfrag(B, ldb, col0 + 16 * j, k0, lane);
#pragma unroll
      for (int i = 0; i < 4; ++i) acc[i][j] = wmma_bf16(a[i], b, acc[i][j]);
    }
  }
}

// ---------------- setup kernels ----------------

__global__ __launch_bounds__(256) void k_init_x(const float* x, const float* pos,
                                                float* xc) {
  size_t i = (size_t)blockIdx.x * 256 + threadIdx.x;
  if (i < (size_t)BL_ * D_) xc[i] = x[i] + pos[i % ((size_t)L_ * D_)];
}

// bf16 copy + transposed bf16 copy of a [rows, cols] fp32 matrix
__global__ __launch_bounds__(256) void k_conv(const float* src, bf16_t* w,
                                              bf16_t* wt, int rows, int cols) {
  int i = blockIdx.x * 256 + threadIdx.x;
  if (i < rows * cols) {
    int r = i / cols, c = i % cols;
    float v = src[i];
    w[i] = (bf16_t)v;
    wt[(size_t)c * rows + r] = (bf16_t)v;
  }
}

// ---------------- per-iteration kernels ----------------

// LayerNorm: one block per row of xc; writes bf16 g; block 0 zeroes hop energy.
__global__ __launch_bounds__(256) void k_ln(const float* xc, const float* w,
                                            const float* bia, bf16_t* g,
                                            float* hopE) {
  int row = blockIdx.x, t = threadIdx.x;
  if (row == 0 && t < NB_) hopE[t] = 0.f;
  __shared__ float red[256];
  const float* xr = xc + (size_t)row * D_;
  float v0 = xr[t], v1 = xr[t + 256], v2 = xr[t + 512];
  red[t] = v0 + v1 + v2;
  __syncthreads();
  for (int s = 128; s > 0; s >>= 1) { if (t < s) red[t] += red[t + s]; __syncthreads(); }
  float mu = red[0] * (1.0f / D_);
  __syncthreads();
  float d0 = v0 - mu, d1 = v1 - mu, d2 = v2 - mu;
  red[t] = d0 * d0 + d1 * d1 + d2 * d2;
  __syncthreads();
  for (int s = 128; s > 0; s >>= 1) { if (t < s) red[t] += red[t + s]; __syncthreads(); }
  float rstd = rsqrtf(red[0] * (1.0f / D_) + EPS_);
  bf16_t* gr = g + (size_t)row * D_;
  gr[t]       = (bf16_t)(d0 * rstd * w[t]       + bia[t]);
  gr[t + 256] = (bf16_t)(d1 * rstd * w[t + 256] + bia[t + 256]);
  gr[t + 512] = (bf16_t)(d2 * rstd * w[t + 512] + bia[t + 512]);
}

// Q/K projection: out = g @ W^T (W given [out,in]); optional per-head beta scale.
// Also writes per-head transposed copy outT[b,h,z,l] for later B-operand use.
__global__ __launch_bounds__(256) void k_proj(const bf16_t* g, const bf16_t* W,
                                              const float* beta, int do_scale,
                                              bf16_t* out, bf16_t* outT) {
  int lane = threadIdx.x & 31;
  int wid = blockIdx.x * 8 + (threadIdx.x >> 5);      // 768 tiles
  int row0 = (wid / (D_ / 64)) * 64;
  int col0 = (wid % (D_ / 64)) * 64;
  v8f acc[4][4];
#pragma unroll
  for (int i = 0; i < 4; ++i)
#pragma unroll
    for (int j = 0; j < 4; ++j) zero8(acc[i][j]);
  gemm64(g, D_, W, D_, row0, col0, D_, lane, acc);
  int h = col0 >> 6;                                   // 64-wide tile == one head
  float sc = do_scale ? beta[h] : 1.0f;
  int n = lane & 15, hi = lane >> 4;
  int b = row0 >> 10, l0 = row0 & 1023;
  bf16_t* tbase = outT + (size_t)(b * H_ + h) * HEAD_ * L_;
#pragma unroll
  for (int i = 0; i < 4; ++i)
#pragma unroll
    for (int j = 0; j < 4; ++j) {
      v8bf t;
#pragma unroll
      for (int r = 0; r < 8; ++r) {
        float v = acc[i][j][r] * sc;
        t[r] = (bf16_t)v;
        out[(size_t)(row0 + 16 * i + hi * 8 + r) * D_ + col0 + 16 * j + n] = (bf16_t)v;
      }
      *(v8bf*)(tbase + (size_t)(16 * j + n) * L_ + l0 + 16 * i + hi * 8) = t;
    }
}

// Pass 1: lse[b,h,q] = logsumexp_k (Qs_q . K_k)   (beta already folded into Qs)
__global__ __launch_bounds__(256) void k_lse(const bf16_t* qs, const bf16_t* kb,
                                             float* lse) {
  int lane = threadIdx.x & 31;
  int wid = blockIdx.x * 8 + (threadIdx.x >> 5);       // 3072 waves
  int q0 = (wid & 63) * 16;
  int bh = wid >> 6;
  int b = bh / H_, h = bh % H_;
  int hi = lane >> 4;
  const bf16_t* Qrow = qs + (size_t)(b << 10) * D_ + h * HEAD_;
  const bf16_t* Krow = kb + (size_t)(b << 10) * D_ + h * HEAD_;
  v16bf a0 = load_afrag(Qrow, D_, q0, 0, lane);
  v16bf a1 = load_afrag(Qrow, D_, q0, 32, lane);
  float m[8], s[8];
#pragma unroll
  for (int r = 0; r < 8; ++r) { m[r] = -3.4e38f; s[r] = 0.f; }
  for (int n0 = 0; n0 < L_; n0 += 16) {
    v16bf b0 = load_bfrag(Krow, D_, n0, 0, lane);
    v16bf b1 = load_bfrag(Krow, D_, n0, 32, lane);
    v8f c; zero8(c);
    c = wmma_bf16(a0, b0, c);
    c = wmma_bf16(a1, b1, c);
#pragma unroll
    for (int r = 0; r < 8; ++r) {
      float v = c[r];
      float t = v;
      t = fmaxf(t, __shfl_xor(t, 1, 16));
      t = fmaxf(t, __shfl_xor(t, 2, 16));
      t = fmaxf(t, __shfl_xor(t, 4, 16));
      t = fmaxf(t, __shfl_xor(t, 8, 16));
      float mn = fmaxf(m[r], t);
      float p = __expf(v - mn);
      p += __shfl_xor(p, 1, 16);
      p += __shfl_xor(p, 2, 16);
      p += __shfl_xor(p, 4, 16);
      p += __shfl_xor(p, 8, 16);
      s[r] = s[r] * __expf(m[r] - mn) + p;
      m[r] = mn;
    }
  }
  if ((lane & 15) == 0) {
#pragma unroll
    for (int r = 0; r < 8; ++r)
      lse[((size_t)bh << 10) + q0 + hi * 8 + r] = m[r] + __logf(s[r]);
  }
}

// Pass 2a: Gq = P @ K  (per b,h; wave owns 16 q rows).  P relayout via LDS.
__global__ __launch_bounds__(256) void k_gq(const bf16_t* qs, const bf16_t* kb,
                                            const bf16_t* kt, const float* lse,
                                            bf16_t* gq) {
  __shared__ bf16_t Pt[8][16 * 72];
  int lane = threadIdx.x & 31;
  int wl = threadIdx.x >> 5;
  int wid = blockIdx.x * 8 + wl;
  int q0 = (wid & 63) * 16;
  int bh = wid >> 6;
  int b = bh / H_, h = bh % H_;
  int n = lane & 15, hi = lane >> 4;
  const bf16_t* Qrow = qs + (size_t)(b << 10) * D_ + h * HEAD_;
  const bf16_t* Krow = kb + (size_t)(b << 10) * D_ + h * HEAD_;
  const bf16_t* KT = kt + (size_t)(b * H_ + h) * HEAD_ * L_;
  v16bf a0 = load_afrag(Qrow, D_, q0, 0, lane);
  v16bf a1 = load_afrag(Qrow, D_, q0, 32, lane);
  float lrow[8];
#pragma unroll
  for (int r = 0; r < 8; ++r) lrow[r] = lse[((size_t)bh << 10) + q0 + hi * 8 + r];
  v8f G[4];
#pragma unroll
  for (int j = 0; j < 4; ++j) zero8(G[j]);
  bf16_t* T = &Pt[wl][0];
  for (int k0 = 0; k0 < L_; k0 += 64) {
#pragma unroll
    for (int kt4 = 0; kt4 < 4; ++kt4) {
      v16bf b0 = load_bfrag(Krow, D_, k0 + kt4 * 16, 0, lane);
      v16bf b1 = load_bfrag(Krow, D_, k0 + kt4 * 16, 32, lane);
      v8f c; zero8(c);
      c = wmma_bf16(a0, b0, c);
      c = wmma_bf16(a1, b1, c);
#pragma unroll
      for (int r = 0; r < 8; ++r)
        T[(hi * 8 + r) * 72 + kt4 * 16 + n] = (bf16_t)__expf(c[r] - lrow[r]);
    }
    asm volatile("s_wait_dscnt 0" ::: "memory");
    v16bf p0 = load_afrag(T, 72, 0, 0, lane);
    v16bf p1 = load_afrag(T, 72, 0, 32, lane);
#pragma unroll
    for (int j = 0; j < 4; ++j) {
      v16bf bz0 = load_bfrag(KT, L_, j * 16, k0, lane);
      v16bf bz1 = load_bfrag(KT, L_, j * 16, k0 + 32, lane);
      G[j] = wmma_bf16(p0, bz0, G[j]);
      G[j] = wmma_bf16(p1, bz1, G[j]);
    }
  }
#pragma unroll
  for (int j = 0; j < 4; ++j)
#pragma unroll
    for (int r = 0; r < 8; ++r)
      gq[(size_t)((b << 10) + q0 + hi * 8 + r) * D_ + h * HEAD_ + j * 16 + n] =
          (bf16_t)G[j][r];
}

// Pass 2b: Gk = (1/beta_h) * P^T @ Qs  (wave owns 16 k rows).
__global__ __launch_bounds__(256) void k_gk(const bf16_t* kb, const bf16_t* qs,
                                            const bf16_t* qt, const float* lse,
                                            const float* beta, bf16_t* gk) {
  __shared__ bf16_t Pt[8][16 * 72];
  int lane = threadIdx.x & 31;
  int wl = threadIdx.x >> 5;
  int wid = blockIdx.x * 8 + wl;
  int k0s = (wid & 63) * 16;
  int bh = wid >> 6;
  int b = bh / H_, h = bh % H_;
  int n = lane & 15, hi = lane >> 4;
  const bf16_t* Krow = kb + (size_t)(b << 10) * D_ + h * HEAD_;
  const bf16_t* Qrow = qs + (size_t)(b << 10) * D_ + h * HEAD_;
  const bf16_t* QT = qt + (size_t)(b * H_ + h) * HEAD_ * L_;
  v16bf a0 = load_afrag(Krow, D_, k0s, 0, lane);
  v16bf a1 = load_afrag(Krow, D_, k0s, 32, lane);
  float invb = 1.0f / beta[h];
  v8f G[4];
#pragma unroll
  for (int j = 0; j < 4; ++j) zero8(G[j]);
  bf16_t* T = &Pt[wl][0];
  for (int q0 = 0; q0 < L_; q0 += 64) {
#pragma unroll
    for (int qt4 = 0; qt4 < 4; ++qt4) {
      v16bf b0 = load_bfrag(Qrow, D_, q0 + qt4 * 16, 0, lane);
      v16bf b1 = load_bfrag(Qrow, D_, q0 + qt4 * 16, 32, lane);
      v8f c; zero8(c);
      c = wmma_bf16(a0, b0, c);   // c[k-row, q-col] = att(q,k)
      c = wmma_bf16(a1, b1, c);
      float lq = lse[((size_t)bh << 10) + q0 + qt4 * 16 + n];
#pragma unroll
      for (int r = 0; r < 8; ++r)
        T[(hi * 8 + r) * 72 + qt4 * 16 + n] = (bf16_t)__expf(c[r] - lq);
    }
    asm volatile("s_wait_dscnt 0" ::: "memory");
    v16bf p0 = load_afrag(T, 72, 0, 0, lane);
    v16bf p1 = load_afrag(T, 72, 0, 32, lane);
#pragma unroll
    for (int j = 0; j < 4; ++j) {
      v16bf bz0 = load_bfrag(QT, L_, j * 16, q0, lane);
      v16bf bz1 = load_bfrag(QT, L_, j * 16, q0 + 32, lane);
      G[j] = wmma_bf16(p0, bz0, G[j]);
      G[j] = wmma_bf16(p1, bz1, G[j]);
    }
  }
#pragma unroll
  for (int j = 0; j < 4; ++j)
#pragma unroll
    for (int r = 0; r < 8; ++r)
      gk[(size_t)((b << 10) + k0s + hi * 8 + r) * D_ + h * HEAD_ + j * 16 + n] =
          (bf16_t)(G[j][r] * invb);
}

// Hopfield forward: H1 = g @ W_hop, rstore = beta*relu(beta*H1), energy atomic.
__global__ __launch_bounds__(256) void k_hop(const bf16_t* g, const bf16_t* WT,
                                             const float* beta_hop, bf16_t* rout,
                                             float* hopE) {
  int lane = threadIdx.x & 31;
  int wid = blockIdx.x * 8 + (threadIdx.x >> 5);       // 3072 tiles
  int row0 = (wid / (HID_ / 64)) * 64;
  int col0 = (wid % (HID_ / 64)) * 64;
  v8f acc[4][4];
#pragma unroll
  for (int i = 0; i < 4; ++i)
#pragma unroll
    for (int j = 0; j < 4; ++j) zero8(acc[i][j]);
  gemm64(g, D_, WT, D_, row0, col0, D_, lane, acc);
  float bh = beta_hop[0];
  int n = lane & 15, hi = lane >> 4;
  float e = 0.f;
#pragma unroll
  for (int i = 0; i < 4; ++i)
#pragma unroll
    for (int j = 0; j < 4; ++j)
#pragma unroll
      for (int r = 0; r < 8; ++r) {
        float hv = bh * acc[i][j][r];
        float rl = fmaxf(hv, 0.f);
        e += rl * rl;
        rout[(size_t)(row0 + 16 * i + hi * 8 + r) * HID_ + col0 + 16 * j + n] =
            (bf16_t)(bh * rl);
      }
  e += __shfl_xor(e, 1, 32);  e += __shfl_xor(e, 2, 32);
  e += __shfl_xor(e, 4, 32);  e += __shfl_xor(e, 8, 32);
  e += __shfl_xor(e, 16, 32);
  if (lane == 0) atomicAdd(&hopE[row0 >> 10], -0.5f * e);
}

// Update: xc += ALPHA * (Gq@Wq + Gk@Wk + r@W_hop^T)   (== xc - ALPHA*grad)
__global__ __launch_bounds__(256) void k_update(const bf16_t* gq, const bf16_t* wqt,
                                                const bf16_t* gk, const bf16_t* wkt,
                                                const bf16_t* rr, const bf16_t* wh,
                                                float* xc) {
  int lane = threadIdx.x & 31;
  int wid = blockIdx.x * 8 + (threadIdx.x >> 5);       // 768 tiles
  int row0 = (wid / (D_ / 64)) * 64;
  int col0 = (wid % (D_ / 64)) * 64;
  v8f acc[4][4];
#pragma unroll
  for (int i = 0; i < 4; ++i)
#pragma unroll
    for (int j = 0; j < 4; ++j) zero8(acc[i][j]);
  gemm64(gq, D_, wqt, D_, row0, col0, D_, lane, acc);
  gemm64(gk, D_, wkt, D_, row0, col0, D_, lane, acc);
  gemm64(rr, HID_, wh, HID_, row0, col0, HID_, lane, acc);
  int n = lane & 15, hi = lane >> 4;
#pragma unroll
  for (int i = 0; i < 4; ++i)
#pragma unroll
    for (int j = 0; j < 4; ++j)
#pragma unroll
      for (int r = 0; r < 8; ++r) {
        size_t idx = (size_t)(row0 + 16 * i + hi * 8 + r) * D_ + col0 + 16 * j + n;
        xc[idx] += ALPHA_ * acc[i][j][r];
      }
}

// energies[it, b] = sum_h -(1/beta_h) * sum_q lse + hopE[b]
__global__ __launch_bounds__(256) void k_energy(const float* lse, const float* beta,
                                                const float* hopE, float* out,
                                                int it) {
  int b = blockIdx.x, t = threadIdx.x;
  __shared__ float red[256];
  float sum = 0.f;
  for (int i = t; i < H_ * L_; i += 256) {
    int h = i >> 10;
    sum -= lse[(size_t)b * H_ * L_ + i] / beta[h];
  }
  red[t] = sum;
  __syncthreads();
  for (int s = 128; s > 0; s >>= 1) { if (t < s) red[t] += red[t + s]; __syncthreads(); }
  if (t == 0) out[(size_t)BL_ * D_ + it * NB_ + b] = red[0] + hopE[b];
}

// ---------------- workspace layout (bytes) ----------------
static constexpr size_t SZA  = (size_t)BL_ * D_ * 2;    // bf16 activation [4096,768]
static constexpr size_t SZW  = (size_t)D_ * D_ * 2;     // bf16 [768,768]
static constexpr size_t SZWH = (size_t)D_ * HID_ * 2;   // bf16 [768,3072]
static constexpr size_t O_G   = 0;
static constexpr size_t O_QS  = O_G  + SZA;
static constexpr size_t O_KB  = O_QS + SZA;
static constexpr size_t O_QT  = O_KB + SZA;
static constexpr size_t O_KT  = O_QT + SZA;
static constexpr size_t O_GQ  = O_KT + SZA;
static constexpr size_t O_GK  = O_GQ + SZA;
static constexpr size_t O_R   = O_GK + SZA;
static constexpr size_t O_WQB = O_R   + (size_t)BL_ * HID_ * 2;
static constexpr size_t O_WKB = O_WQB + SZW;
static constexpr size_t O_WQT = O_WKB + SZW;
static constexpr size_t O_WKT = O_WQT + SZW;
static constexpr size_t O_WHB = O_WKT + SZW;
static constexpr size_t O_WHT = O_WHB + SZWH;
static constexpr size_t O_LSE = O_WHT + SZWH;
static constexpr size_t O_HOPE = O_LSE + (size_t)NB_ * H_ * L_ * 4;

extern "C" void kernel_launch(void* const* d_in, const int* in_sizes, int n_in,
                              void* d_out, int out_size, void* d_ws, size_t ws_size,
                              hipStream_t stream) {
  (void)in_sizes; (void)n_in; (void)out_size; (void)ws_size;
  const float* x    = (const float*)d_in[0];
  const float* pos  = (const float*)d_in[1];
  const float* Wq   = (const float*)d_in[2];
  const float* Wk   = (const float*)d_in[3];
  const float* beta = (const float*)d_in[4];
  const float* Whop = (const float*)d_in[5];
  const float* bhop = (const float*)d_in[6];
  const float* lnw  = (const float*)d_in[7];
  const float* lnb  = (const float*)d_in[8];

  float* xc = (float*)d_out;                 // iterate in-place; final state = output
  float* out_f = (float*)d_out;
  char* ws = (char*)d_ws;
  bf16_t* gbf  = (bf16_t*)(ws + O_G);
  bf16_t* qs   = (bf16_t*)(ws + O_QS);
  bf16_t* kb   = (bf16_t*)(ws + O_KB);
  bf16_t* qt   = (bf16_t*)(ws + O_QT);
  bf16_t* kt   = (bf16_t*)(ws + O_KT);
  bf16_t* gq   = (bf16_t*)(ws + O_GQ);
  bf16_t* gk   = (bf16_t*)(ws + O_GK);
  bf16_t* rr   = (bf16_t*)(ws + O_R);
  bf16_t* wqb  = (bf16_t*)(ws + O_WQB);
  bf16_t* wkb  = (bf16_t*)(ws + O_WKB);
  bf16_t* wqt  = (bf16_t*)(ws + O_WQT);
  bf16_t* wkt  = (bf16_t*)(ws + O_WKT);
  bf16_t* whb  = (bf16_t*)(ws + O_WHB);
  bf16_t* wht  = (bf16_t*)(ws + O_WHT);
  float*  lse  = (float*)(ws + O_LSE);
  float*  hopE = (float*)(ws + O_HOPE);

  k_init_x<<<(BL_ * D_) / 256, 256, 0, stream>>>(x, pos, xc);
  k_conv<<<(D_ * D_) / 256, 256, 0, stream>>>(Wq, wqb, wqt, D_, D_);
  k_conv<<<(D_ * D_) / 256, 256, 0, stream>>>(Wk, wkb, wkt, D_, D_);
  k_conv<<<(D_ * HID_) / 256, 256, 0, stream>>>(Whop, whb, wht, D_, HID_);

  const int GB768  = (BL_ / 64) * (D_ / 64) / 8;    // 96
  const int GB3072 = (BL_ / 64) * (HID_ / 64) / 8;  // 384
  const int GBATT  = NB_ * H_ * (L_ / 16) / 8;      // 384

  for (int it = 0; it < NIT_; ++it) {
    k_ln<<<BL_, 256, 0, stream>>>(xc, lnw, lnb, gbf, hopE);
    k_proj<<<GB768, 256, 0, stream>>>(gbf, wqb, beta, 1, qs, qt);   // Qs = beta*Q
    k_proj<<<GB768, 256, 0, stream>>>(gbf, wkb, beta, 0, kb, kt);   // K
    k_lse<<<GBATT, 256, 0, stream>>>(qs, kb, lse);
    k_gq<<<GBATT, 256, 0, stream>>>(qs, kb, kt, lse, gq);
    k_gk<<<GBATT, 256, 0, stream>>>(kb, qs, qt, lse, beta, gk);
    k_hop<<<GB3072, 256, 0, stream>>>(gbf, wht, bhop, rr, hopE);
    k_update<<<GB768, 256, 0, stream>>>(gq, wqt, gk, wkt, rr, whb, xc);
    k_energy<<<NB_, 256, 0, stream>>>(lse, beta, hopE, out_f, it);
  }
}
